// ByteCombineCNN_3006477107601
// MI455X (gfx1250) — compile-verified, matrix operand order checked by reading, not run on hardware
//
#include <hip/hip_runtime.h>
#include <hip/hip_bf16.h>

typedef __attribute__((ext_vector_type(16))) __bf16 v16bf;
typedef __attribute__((ext_vector_type(8)))  __bf16 v8bf;
typedef __attribute__((ext_vector_type(8)))  float  v8f;

#define N_TOK 32768

// workspace byte offsets (all 256B aligned)
#define OFF_WBIG  0u
#define OFF_HW1   (OFF_WBIG + 448u*512u*2u)          // 458752
#define OFF_HW2   (OFF_HW1 + 224u*128u*2u)           // 516096
#define OFF_PW    (OFF_HW2 + 224u*128u*2u)           // 573440
#define OFF_BIASC (OFF_PW + 512u*128u*2u)            // 704512
#define OFF_H     ((OFF_BIASC + 112u*4u + 255u) & ~255u) // 705024

__device__ __forceinline__ unsigned short f2bf(float f) {
    unsigned int u = __float_as_uint(f);
    u += 0x7FFFu + ((u >> 16) & 1u);   // round-to-nearest-even
    return (unsigned short)(u >> 16);
}

__device__ __forceinline__ v8f wmma_bf16(v16bf a, v16bf b, v8f c) {
    return __builtin_amdgcn_wmma_f32_16x16x32_bf16(
        false, a, false, b, (short)0, c, false, false);
}

// One lane's 16 bf16 elements of a 16x16x32 fragment: chunk0 at k = lhi*8,
// chunk1 at k = 16 + lhi*8 (two 16B contiguous loads).
__device__ __forceinline__ v16bf load_frag(const unsigned short* base, int lhi) {
    v16bf r;
    *reinterpret_cast<v8bf*>(&r)       = *reinterpret_cast<const v8bf*>(base + lhi * 8);
    *(reinterpret_cast<v8bf*>(&r) + 1) = *reinterpret_cast<const v8bf*>(base + 16 + lhi * 8);
    return r;
}

// ---------------------------------------------------------------------------
// Kernel 1: pack weights to bf16 (im2col conv weight matrix + padded GEMM Ws)
// ---------------------------------------------------------------------------
__global__ void prep_kernel(
    const float* cw0, const float* cw1, const float* cw2, const float* cw3,
    const float* cw4, const float* cw5, const float* cw6,
    const float* cb0, const float* cb1, const float* cb2, const float* cb3,
    const float* cb4, const float* cb5, const float* cb6,
    const float* hw1, const float* hw2, const float* pw,
    unsigned short* Wbig, unsigned short* hw1b, unsigned short* hw2b,
    unsigned short* pwb, float* biasC)
{
    const float* cw[7] = {cw0, cw1, cw2, cw3, cw4, cw5, cw6};
    const float* cb[7] = {cb0, cb1, cb2, cb3, cb4, cb5, cb6};
    const int NW = 448 * 512, NH = 224 * 128, NP = 512 * 128;
    const int total = NW + 2 * NH + NP + 112;
    for (int idx = blockIdx.x * blockDim.x + threadIdx.x; idx < total;
         idx += gridDim.x * blockDim.x) {
        if (idx < NW) {
            int col = idx >> 9, k = idx & 511;
            int fi = (col < 32) ? 0 : (col < 88) ? 1 : (col < 160) ? 2 :
                     (col < 240) ? 3 : (col < 320) ? 4 : (col < 392) ? 5 : 6;
            const int colbase[7] = {0, 32, 88, 160, 240, 320, 392};
            int w = fi + 1, c = 4 * (fi + 1);
            int rel = col - colbase[fi];
            int p = rel / c, ch = rel - p * c;
            int t8 = k >> 6, e = k & 63;
            int kk = t8 - p;
            float v = 0.0f;
            if (kk >= 0 && kk < w) v = cw[fi][(ch * 64 + e) * w + kk];
            Wbig[col * 512 + k] = f2bf(v);
        } else if (idx < NW + 2 * NH) {
            int j = idx - NW;
            const float* src = hw1;
            unsigned short* dst = hw1b;
            if (j >= NH) { j -= NH; src = hw2; dst = hw2b; }
            int row = j >> 7, kk = j & 127;
            float v = (kk < 112) ? src[row * 112 + kk] : 0.0f;
            dst[row * 128 + kk] = f2bf(v);
        } else if (idx < NW + 2 * NH + NP) {
            int j = idx - NW - 2 * NH;
            int row = j >> 7, kk = j & 127;
            float v = (kk < 112) ? pw[row * 112 + kk] : 0.0f;
            pwb[row * 128 + kk] = f2bf(v);
        } else {
            int oc = idx - NW - 2 * NH - NP;
            int fi = (oc < 4) ? 0 : (oc < 12) ? 1 : (oc < 24) ? 2 :
                     (oc < 40) ? 3 : (oc < 60) ? 4 : (oc < 84) ? 5 : 6;
            const int outbase[7] = {0, 4, 12, 24, 40, 60, 84};
            biasC[oc] = cb[fi][oc - outbase[fi]];
        }
    }
}

// ---------------------------------------------------------------------------
// Kernel 2: im2col conv GEMM [32tok x 512] x [512 x 448] + max/bias/relu -> h
// 32 tokens/WG (2 M-tiles per wave), B double-buffered in registers.
// LDS: 32KB union of A-stage (bf16 32x512) and acc-stage (f32 16x448 halves).
// ---------------------------------------------------------------------------
__global__ __launch_bounds__(256) void conv_gemm_kernel(
    const float* __restrict__ features, const unsigned short* __restrict__ Wbig,
    const float* __restrict__ biasC, float* __restrict__ h)
{
    __shared__ alignas(32) unsigned char smemRaw[32 * 512 * 2];
    unsigned short* ldsA = (unsigned short*)smemRaw;
    float* accbuf = (float*)smemRaw;          // overlays ldsA after GEMM phase
    const int tid = threadIdx.x;
    const int tokBase = blockIdx.x * 32;

    // Stage A: 32 tokens x 512 f32 -> bf16 in LDS
    const float4* src = (const float4*)(features + (size_t)tokBase * 512);
    for (int i = tid; i < 32 * 512 / 4; i += 256) {
        float4 v = src[i];
        int e = i * 4;
        ldsA[e + 0] = f2bf(v.x); ldsA[e + 1] = f2bf(v.y);
        ldsA[e + 2] = f2bf(v.z); ldsA[e + 3] = f2bf(v.w);
    }
    __syncthreads();

    const int wv = tid >> 5, lane = tid & 31;
    const int lrow = lane & 15, lhi = lane >> 4;
    v8f a00 = {}, a01 = {}, a02 = {}, a03 = {};   // M rows 0..15, 4 col tiles
    v8f a10 = {}, a11 = {}, a12 = {}, a13 = {};   // M rows 16..31

    if (wv < 7) {                      // wave-uniform: EXEC stays all-ones
        const int ct0 = wv * 4;        // 7 waves x 4 col tiles = 28 (448 cols)
        const unsigned short* ap0 = ldsA + lrow * 512;
        const unsigned short* ap1 = ldsA + (16 + lrow) * 512;
        const unsigned short* bp0 = Wbig + ((ct0 + 0) * 16 + lrow) * 512;
        const unsigned short* bp1 = Wbig + ((ct0 + 1) * 16 + lrow) * 512;
        const unsigned short* bp2 = Wbig + ((ct0 + 2) * 16 + lrow) * 512;
        const unsigned short* bp3 = Wbig + ((ct0 + 3) * 16 + lrow) * 512;
        v16bf b0 = load_frag(bp0, lhi);
        v16bf b1 = load_frag(bp1, lhi);
        v16bf b2 = load_frag(bp2, lhi);
        v16bf b3 = load_frag(bp3, lhi);
        for (int ks = 0; ks < 16; ++ks) {
            const int ko = ks * 32, kn = ko + 32;
            v16bf x0 = load_frag(ap0 + ko, lhi);
            v16bf x1 = load_frag(ap1 + ko, lhi);
            // prefetch next k-step B (last iter overreads 64B into next ws
            // region -- allocated, unused, deterministic)
            v16bf nb0 = load_frag(bp0 + kn, lhi);
            v16bf nb1 = load_frag(bp1 + kn, lhi);
            v16bf nb2 = load_frag(bp2 + kn, lhi);
            v16bf nb3 = load_frag(bp3 + kn, lhi);
            a00 = wmma_bf16(x0, b0, a00);  a10 = wmma_bf16(x1, b0, a10);
            a01 = wmma_bf16(x0, b1, a01);  a11 = wmma_bf16(x1, b1, a11);
            a02 = wmma_bf16(x0, b2, a02);  a12 = wmma_bf16(x1, b2, a12);
            a03 = wmma_bf16(x0, b3, a03);  a13 = wmma_bf16(x1, b3, a13);
            b0 = nb0; b1 = nb1; b2 = nb2; b3 = nb3;
        }
    }
    __syncthreads();   // ldsA dead; accbuf phase begins

    // ---- half 0: M rows 0..15 ----
    if (wv < 7) {
        const int ct0 = wv * 4;
        for (int r = 0; r < 8; ++r) {
            const int row = r + 8 * lhi;
            accbuf[row * 448 + (ct0 + 0) * 16 + lrow] = a00[r];
            accbuf[row * 448 + (ct0 + 1) * 16 + lrow] = a01[r];
            accbuf[row * 448 + (ct0 + 2) * 16 + lrow] = a02[r];
            accbuf[row * 448 + (ct0 + 3) * 16 + lrow] = a03[r];
        }
    }
    __syncthreads();
    for (int idx = tid; idx < 16 * 112; idx += 256) {
        int row = idx / 112, oc = idx - row * 112;
        int fi = (oc < 4) ? 0 : (oc < 12) ? 1 : (oc < 24) ? 2 :
                 (oc < 40) ? 3 : (oc < 60) ? 4 : (oc < 84) ? 5 : 6;
        const int outbase[7] = {0, 4, 12, 24, 40, 60, 84};
        const int colbase[7] = {0, 32, 88, 160, 240, 320, 392};
        int c = 4 * (fi + 1), npos = 8 - fi;
        int ch = oc - outbase[fi];
        const float* arow = &accbuf[row * 448 + colbase[fi] + ch];
        float m = -3.402823e38f;
        for (int p = 0; p < npos; ++p) m = fmaxf(m, arow[p * c]);
        m = fmaxf(m + biasC[oc], 0.0f);
        h[(size_t)(tokBase + row) * 112 + oc] = m;
    }
    __syncthreads();

    // ---- half 1: M rows 16..31 ----
    if (wv < 7) {
        const int ct0 = wv * 4;
        for (int r = 0; r < 8; ++r) {
            const int row = r + 8 * lhi;
            accbuf[row * 448 + (ct0 + 0) * 16 + lrow] = a10[r];
            accbuf[row * 448 + (ct0 + 1) * 16 + lrow] = a11[r];
            accbuf[row * 448 + (ct0 + 2) * 16 + lrow] = a12[r];
            accbuf[row * 448 + (ct0 + 3) * 16 + lrow] = a13[r];
        }
    }
    __syncthreads();
    for (int idx = tid; idx < 16 * 112; idx += 256) {
        int row = idx / 112, oc = idx - row * 112;
        int fi = (oc < 4) ? 0 : (oc < 12) ? 1 : (oc < 24) ? 2 :
                 (oc < 40) ? 3 : (oc < 60) ? 4 : (oc < 84) ? 5 : 6;
        const int outbase[7] = {0, 4, 12, 24, 40, 60, 84};
        const int colbase[7] = {0, 32, 88, 160, 240, 320, 392};
        int c = 4 * (fi + 1), npos = 8 - fi;
        int ch = oc - outbase[fi];
        const float* arow = &accbuf[row * 448 + colbase[fi] + ch];
        float m = -3.402823e38f;
        for (int p = 0; p < npos; ++p) m = fmaxf(m, arow[p * c]);
        m = fmaxf(m + biasC[oc], 0.0f);
        h[(size_t)(tokBase + 16 + row) * 112 + oc] = m;
    }
}

// ---------------------------------------------------------------------------
// Kernel 3: fused highway x2 + projection. 32 tokens/WG; h loaded once,
// both highway layers iterate in LDS, projection streams its B panel from L2
// and stores directly to out.
// ---------------------------------------------------------------------------
__global__ __launch_bounds__(256) void head_kernel(
    const float* __restrict__ h,
    const unsigned short* __restrict__ hw1b, const float* __restrict__ hb1,
    const unsigned short* __restrict__ hw2b, const float* __restrict__ hb2,
    const unsigned short* __restrict__ pwb, const float* __restrict__ pb,
    float* __restrict__ out)
{
    __shared__ alignas(32) unsigned short ldsA[32 * 128];  // bf16 x, k-padded
    __shared__ float ldsX[32 * 112];                       // f32 x (residual)
    __shared__ float ldsP[32 * 224];                       // highway pre-acts
    const int tid = threadIdx.x;
    const int tokBase = blockIdx.x * 32;

    for (int idx = tid; idx < 32 * 128; idx += 256) {
        int row = idx >> 7, kk = idx & 127;
        float v = 0.0f;
        if (kk < 112) {
            v = h[(size_t)(tokBase + row) * 112 + kk];
            ldsX[row * 112 + kk] = v;
        }
        ldsA[idx] = f2bf(v);          // pad cols 112..127 stay zero forever
    }
    __syncthreads();

    const int wv = tid >> 5, lane = tid & 31;
    const int lrow = lane & 15, lhi = lane >> 4;

    for (int layer = 0; layer < 2; ++layer) {
        const unsigned short* wb = layer ? hw2b : hw1b;
        const float* bias = layer ? hb2 : hb1;
        if (wv < 7) {                  // 7 waves x 2 col tiles = 14 (224 cols)
            const int ct0 = wv * 2;
            v8f a00 = {}, a01 = {}, a10 = {}, a11 = {};
            const unsigned short* ap0 = ldsA + lrow * 128;
            const unsigned short* ap1 = ldsA + (16 + lrow) * 128;
            const unsigned short* b0p = wb + ((ct0 + 0) * 16 + lrow) * 128;
            const unsigned short* b1p = wb + ((ct0 + 1) * 16 + lrow) * 128;
            for (int ks = 0; ks < 4; ++ks) {
                const int ko = ks * 32;
                v16bf x0 = load_frag(ap0 + ko, lhi);
                v16bf x1 = load_frag(ap1 + ko, lhi);
                v16bf bb0 = load_frag(b0p + ko, lhi);
                v16bf bb1 = load_frag(b1p + ko, lhi);
                a00 = wmma_bf16(x0, bb0, a00);  a10 = wmma_bf16(x1, bb0, a10);
                a01 = wmma_bf16(x0, bb1, a01);  a11 = wmma_bf16(x1, bb1, a11);
            }
            for (int r = 0; r < 8; ++r) {
                const int row = r + 8 * lhi;
                ldsP[row * 224 + (ct0 + 0) * 16 + lrow] = a00[r];
                ldsP[row * 224 + (ct0 + 1) * 16 + lrow] = a01[r];
                ldsP[(16 + row) * 224 + (ct0 + 0) * 16 + lrow] = a10[r];
                ldsP[(16 + row) * 224 + (ct0 + 1) * 16 + lrow] = a11[r];
            }
        }
        __syncthreads();
        for (int idx = tid; idx < 32 * 112; idx += 256) {
            int row = idx / 112, oc = idx - row * 112;
            float proj = ldsP[row * 224 + oc] + bias[oc];
            float gate = ldsP[row * 224 + 112 + oc] + bias[112 + oc];
            float g = 1.0f / (1.0f + __expf(-gate));
            float x = ldsX[row * 112 + oc];
            float nx = g * x + (1.0f - g) * fmaxf(proj, 0.0f);
            ldsX[row * 112 + oc] = nx;
            ldsA[row * 128 + oc] = f2bf(nx);
        }
        __syncthreads();
    }

    // projection: 8 waves x 4 col tiles = 32 (512 cols), 2 M-tiles per wave
    {
        const int ct0 = wv * 4;
        v8f a00 = {}, a01 = {}, a02 = {}, a03 = {};
        v8f a10 = {}, a11 = {}, a12 = {}, a13 = {};
        const unsigned short* ap0 = ldsA + lrow * 128;
        const unsigned short* ap1 = ldsA + (16 + lrow) * 128;
        const unsigned short* b0p = pwb + ((ct0 + 0) * 16 + lrow) * 128;
        const unsigned short* b1p = pwb + ((ct0 + 1) * 16 + lrow) * 128;
        const unsigned short* b2p = pwb + ((ct0 + 2) * 16 + lrow) * 128;
        const unsigned short* b3p = pwb + ((ct0 + 3) * 16 + lrow) * 128;
        for (int ks = 0; ks < 4; ++ks) {
            const int ko = ks * 32;
            v16bf x0 = load_frag(ap0 + ko, lhi);
            v16bf x1 = load_frag(ap1 + ko, lhi);
            v16bf bb0 = load_frag(b0p + ko, lhi);
            v16bf bb1 = load_frag(b1p + ko, lhi);
            v16bf bb2 = load_frag(b2p + ko, lhi);
            v16bf bb3 = load_frag(b3p + ko, lhi);
            a00 = wmma_bf16(x0, bb0, a00);  a10 = wmma_bf16(x1, bb0, a10);
            a01 = wmma_bf16(x0, bb1, a01);  a11 = wmma_bf16(x1, bb1, a11);
            a02 = wmma_bf16(x0, bb2, a02);  a12 = wmma_bf16(x1, bb2, a12);
            a03 = wmma_bf16(x0, bb3, a03);  a13 = wmma_bf16(x1, bb3, a13);
        }
        for (int r = 0; r < 8; ++r) {
            const int t0 = tokBase + r + 8 * lhi;
            const int t1 = t0 + 16;
            float* o0 = out + (size_t)t0 * 512;
            float* o1 = out + (size_t)t1 * 512;
            int c0 = (ct0 + 0) * 16 + lrow;
            int c1 = (ct0 + 1) * 16 + lrow;
            int c2 = (ct0 + 2) * 16 + lrow;
            int c3 = (ct0 + 3) * 16 + lrow;
            o0[c0] = a00[r] + pb[c0];  o1[c0] = a10[r] + pb[c0];
            o0[c1] = a01[r] + pb[c1];  o1[c1] = a11[r] + pb[c1];
            o0[c2] = a02[r] + pb[c2];  o1[c2] = a12[r] + pb[c2];
            o0[c3] = a03[r] + pb[c3];  o1[c3] = a13[r] + pb[c3];
        }
    }
}

// ---------------------------------------------------------------------------
extern "C" void kernel_launch(void* const* d_in, const int* in_sizes, int n_in,
                              void* d_out, int out_size, void* d_ws, size_t ws_size,
                              hipStream_t stream) {
    (void)in_sizes; (void)n_in; (void)out_size; (void)ws_size;
    const float* features = (const float*)d_in[0];
    const float* cw[7]; const float* cb[7];
    for (int i = 0; i < 7; ++i) {
        cw[i] = (const float*)d_in[1 + 2 * i];
        cb[i] = (const float*)d_in[2 + 2 * i];
    }
    const float* hw_w1  = (const float*)d_in[15];
    const float* hw_b1  = (const float*)d_in[16];
    const float* hw_w2  = (const float*)d_in[17];
    const float* hw_b2  = (const float*)d_in[18];
    const float* proj_w = (const float*)d_in[19];
    const float* proj_b = (const float*)d_in[20];

    char* ws = (char*)d_ws;
    unsigned short* Wbig  = (unsigned short*)(ws + OFF_WBIG);
    unsigned short* hw1b  = (unsigned short*)(ws + OFF_HW1);
    unsigned short* hw2b  = (unsigned short*)(ws + OFF_HW2);
    unsigned short* pwb   = (unsigned short*)(ws + OFF_PW);
    float*          biasC = (float*)(ws + OFF_BIASC);
    float*          h     = (float*)(ws + OFF_H);

    prep_kernel<<<1024, 256, 0, stream>>>(
        cw[0], cw[1], cw[2], cw[3], cw[4], cw[5], cw[6],
        cb[0], cb[1], cb[2], cb[3], cb[4], cb[5], cb[6],
        hw_w1, hw_w2, proj_w, Wbig, hw1b, hw2b, pwb, biasC);

    const int nWG = N_TOK / 32;  // 1024
    conv_gemm_kernel<<<nWG, 256, 0, stream>>>(features, Wbig, biasC, h);
    head_kernel<<<nWG, 256, 0, stream>>>(h, hw1b, hw_b1, hw2b, hw_b2,
                                         pwb, proj_b, (float*)d_out);
}
